// RNNModel_31576599560836
// MI455X (gfx1250) — matrix operand order repaired and checked by reading, not compiled, not run on hardware
//
#include <hip/hip_runtime.h>
#include <hip/hip_bf16.h>
#include <math.h>

// ---------------------------------------------------------------------------
// Fused 5-layer tanh RNN (B=262144, F=5, T=4, H=128) + linear head, MI455X.
// Compute-bound (~0.31 TFLOP vs ~22 MB HBM) -> all GEMMs on
// v_wmma_f32_16x16x32_f16, h round-trips through LDS as f16.
// Round-2 fixes vs round-1:
//   * all LDS addressing via integer offsets off the single extern-shared
//     base (compile-time strides) so clang infers addrspace(3) -> ds_load_b128
//     instead of flat_load_b128
//   * layer0 / t==0 peeled via templates -> straight-line code, A-fragments
//     batch-loaded before the WMMA chain (one wait per group)
//   * tanh via hardware transcendentals: 1 - 2/(1+exp(2x))
// ---------------------------------------------------------------------------

typedef __attribute__((ext_vector_type(16))) _Float16 v16h;
typedef __attribute__((ext_vector_type(8)))  _Float16 v8h;
typedef __attribute__((ext_vector_type(8)))  float    v8f;

#define RNN_B 262144
#define RNN_F 5
#define RNN_T 4
#define RNN_H 128
#define RNN_L 5
#define BT    64            // batch rows per workgroup (4 WMMA M-subtiles)

// d_ws layout (in _Float16 units, f32 bias block appended)
#define OFF_W0    0         // layer0 W_ih padded: [128][32]
#define OFF_WHH0  4096      // layer0 W_hh:        [128][128]
#define OFF_WL    20480     // layers 1..4: per layer 32768 halves (W_ih,W_hh)
#define NHALF     151552
#define NBIAS     (RNN_L * RNN_H)

// LDS layout (half units)
#define X0_OFF    0                         // [T][BT][32]
#define A_OFF     (RNN_T * BT * 32)         // [T][BT][128]
#define B_OFF     (A_OFF + RNN_T * BT * RNN_H)
#define SMEM_HALVES (B_OFF + RNN_T * BT * RNN_H)

// ---------------------------------------------------------------------------
// Weight prep: f32 -> f16 image in d_ws (runs every launch; deterministic).
// ---------------------------------------------------------------------------
__global__ void rnn_prep(const float* __restrict__ Wih0,
                         const float* __restrict__ Whh0,
                         const float* __restrict__ bih0,
                         const float* __restrict__ bhh0,
                         const float* __restrict__ Wih,
                         const float* __restrict__ Whh,
                         const float* __restrict__ bih,
                         const float* __restrict__ bhh,
                         _Float16* __restrict__ wsh,
                         float* __restrict__ wsb) {
  int i = blockIdx.x * blockDim.x + threadIdx.x;
  if (i < 4096) {                       // layer0 W_ih, K padded 5 -> 32
    int n = i >> 5, k = i & 31;
    wsh[i] = (_Float16)((k < RNN_F) ? Wih0[n * RNN_F + k] : 0.0f);
    return;
  }
  if (i < OFF_WL) {                     // layer0 W_hh
    wsh[i] = (_Float16)Whh0[i - OFF_WHH0];
    return;
  }
  if (i < NHALF) {                      // layers 1..4
    int j = i - OFF_WL;
    int l = j >> 15;
    int r = j & 32767;
    float v = (r < 16384) ? Wih[l * 16384 + r] : Whh[l * 16384 + (r - 16384)];
    wsh[i] = (_Float16)v;
    return;
  }
  if (i < NHALF + NBIAS) {              // summed biases, f32
    int j = i - NHALF;
    int l = j >> 7, n = j & 127;
    wsb[j] = (l == 0) ? (bih0[n] + bhh0[n])
                      : (bih[(l - 1) * RNN_H + n] + bhh[(l - 1) * RNN_H + n]);
  }
}

// ---------------------------------------------------------------------------
// Fragment loaders (wave32; contiguous k order, same for A and B).
// A 16x32 f16: lane L row = L%16; halves 0..15 = K[(L/16)*16 + 0..15]
// B 32x16 f16: lane L col = L%16; halves 0..15 = K[(L/16)*16 + 0..15]
// ---------------------------------------------------------------------------
__device__ __forceinline__ v16h load_a_frag(const _Float16* p) {
  v8h lo = *(const v8h*)p;
  v8h hi = *(const v8h*)(p + 8);
  v16h r;
#pragma unroll
  for (int i = 0; i < 8; ++i) { r[i] = lo[i]; r[i + 8] = hi[i]; }
  return r;
}

__device__ __forceinline__ v16h load_b_frag(const _Float16* p) {
  v8h lo = *(const v8h*)p;
  v8h hi = *(const v8h*)(p + 8);
  v16h r;
#pragma unroll
  for (int i = 0; i < 8; ++i) { r[i] = lo[i]; r[i + 8] = hi[i]; }
  return r;
}

__device__ __forceinline__ v8f wmma_f16(v16h a, v16h b, v8f c) {
  return __builtin_amdgcn_wmma_f32_16x16x32_f16(false, a, false, b,
                                                (short)0, c, false, false);
}

// fast tanh on hardware transcendentals: tanh(x) = 1 - 2/(1+e^{2x})
__device__ __forceinline__ float fast_tanh(float x) {
  float e = __expf(2.0f * x);
  return 1.0f - 2.0f * __frcp_rn(e + 1.0f);
}

// ---------------------------------------------------------------------------
// One timestep over the whole 64-row tile. All LDS pointers must be derived
// from the extern-shared base by the (inlined) caller. Compile-time D, REC.
// ---------------------------------------------------------------------------
template <int NKC_IN, bool REC>
__device__ __forceinline__ void rnn_step(
    const _Float16* sm_in,    // smem + off: [BT][D] slice for this t
    const _Float16* sm_hprev, // smem + off: [BT][128] slice (REC only)
    _Float16* sm_out,         // smem + off: [BT][128] slice for this t
    const v16h* bih_fr, const v16h* bhh_fr, float bval,
    int l15, int khi, int n0) {
  constexpr int D = NKC_IN * 32;
#pragma unroll
  for (int ms = 0; ms < BT / 16; ++ms) {
    const int m0 = ms * 16;
    // batch all A-fragment ds loads first, then chain the WMMAs
    v16h a_in[NKC_IN];
    const _Float16* ain = sm_in + (m0 + l15) * D + khi * 16;
#pragma unroll
    for (int kc = 0; kc < NKC_IN; ++kc) a_in[kc] = load_a_frag(ain + kc * 32);
    v16h a_h[4];
    if (REC) {
      const _Float16* ah = sm_hprev + (m0 + l15) * RNN_H + khi * 16;
#pragma unroll
      for (int kc = 0; kc < 4; ++kc) a_h[kc] = load_a_frag(ah + kc * 32);
    }
    v8f c = {};
#pragma unroll
    for (int kc = 0; kc < NKC_IN; ++kc) c = wmma_f16(a_in[kc], bih_fr[kc], c);
    if (REC) {
#pragma unroll
      for (int kc = 0; kc < 4; ++kc) c = wmma_f16(a_h[kc], bhh_fr[kc], c);
    }
#pragma unroll
    for (int vi = 0; vi < 8; ++vi) {
      int row = vi + (khi << 3);
      float hv = fast_tanh(c[vi] + bval);
      sm_out[(m0 + row) * RNN_H + n0 + l15] = (_Float16)hv;
    }
  }
}

// One full layer (T steps, t=0 peeled: h_0 = 0 so no recurrent GEMM).
template <int NKC_IN>
__device__ __forceinline__ void rnn_layer(
    _Float16* smem, int in_off, int out_off,
    const _Float16* __restrict__ Wih_g, const _Float16* __restrict__ Whh_g,
    float bval, int l15, int khi, int n0) {
  constexpr int D = NKC_IN * 32;
  // per-wave weight fragments in VGPRs, reused across T x 4 M-subtiles
  v16h bih_fr[NKC_IN], bhh_fr[4];
#pragma unroll
  for (int kc = 0; kc < NKC_IN; ++kc)
    bih_fr[kc] = load_b_frag(Wih_g + (size_t)(n0 + l15) * D + kc * 32 + khi * 16);
#pragma unroll
  for (int kc = 0; kc < 4; ++kc)
    bhh_fr[kc] =
        load_b_frag(Whh_g + (size_t)(n0 + l15) * RNN_H + kc * 32 + khi * 16);

  rnn_step<NKC_IN, false>(smem + in_off, (const _Float16*)0, smem + out_off,
                          bih_fr, bhh_fr, bval, l15, khi, n0);
  __syncthreads();
  for (int t = 1; t < RNN_T; ++t) {
    rnn_step<NKC_IN, true>(smem + in_off + t * BT * D,
                           smem + out_off + (t - 1) * BT * RNN_H,
                           smem + out_off + t * BT * RNN_H,
                           bih_fr, bhh_fr, bval, l15, khi, n0);
    __syncthreads();
  }
}

// ---------------------------------------------------------------------------
// Main fused kernel. Dynamic LDS: 144 KB.
// ---------------------------------------------------------------------------
__global__ __launch_bounds__(256, 1) void rnn_fused(
    const float* __restrict__ x, const _Float16* __restrict__ wsh,
    const float* __restrict__ wsb, const float* __restrict__ Wfc,
    const float* __restrict__ bfc, float* __restrict__ out) {
  extern __shared__ _Float16 smem[];

  const int tid  = threadIdx.x;
  const int wave = tid >> 5;
  const int lane = tid & 31;
  const int n0   = wave << 4;   // this wave's hidden-column tile
  const int l15  = lane & 15;
  const int khi  = lane >> 4;
  const long b0  = (long)blockIdx.x * BT;

  // Warm the weight image (global_prefetch_b8), 4 KB strides.
  if (tid < 38) __builtin_prefetch(wsh + (size_t)tid * 2048, 0, 3);

  // Stage x tile, transposing (B,F,T) -> [t][m][k], K padded 5 -> 32.
  for (int idx = tid; idx < RNN_T * BT * 32; idx += 256) {
    int k = idx & 31;
    int m = (idx >> 5) & (BT - 1);
    int t = idx >> 11;
    float v = (k < RNN_F)
                  ? x[(b0 + m) * (RNN_F * RNN_T) + (size_t)k * RNN_T + t]
                  : 0.0f;
    smem[X0_OFF + idx] = (_Float16)v;
  }
  __syncthreads();

  // Layer 0: D = 32 (padded F), reads x0, writes buffer A.
  rnn_layer<1>(smem, X0_OFF, A_OFF, wsh + OFF_W0, wsh + OFF_WHH0,
               wsb[n0 + l15], l15, khi, n0);

  // Layers 1..4: D = 128, ping-pong A <-> B via integer offsets.
  for (int layer = 1; layer < RNN_L; ++layer) {
    const int in_off  = (layer & 1) ? A_OFF : B_OFF;
    const int out_off = (layer & 1) ? B_OFF : A_OFF;
    const _Float16* Wih_g = wsh + OFF_WL + (size_t)(layer - 1) * 32768;
    const _Float16* Whh_g = Wih_g + 16384;
    rnn_layer<4>(smem, in_off, out_off, Wih_g, Whh_g,
                 wsb[layer * RNN_H + n0 + l15], l15, khi, n0);
  }

  // Head: out[b] = dot(h_T, Wfc) + bfc. Layer 4 wrote buffer A; h_T at t=3.
  const _Float16* hfin = smem + A_OFF + 3 * BT * RNN_H;
  int row = tid >> 2;   // 64 rows, 4 lanes per row
  int sub = tid & 3;
  float acc = 0.0f;
  const _Float16* hp = hfin + row * RNN_H + sub * 32;
  const float* wp = Wfc + sub * 32;
#pragma unroll
  for (int k = 0; k < 32; ++k) acc += (float)hp[k] * wp[k];
  acc += __shfl_xor(acc, 1, 32);
  acc += __shfl_xor(acc, 2, 32);
  if (sub == 0) out[b0 + row] = acc + bfc[0];
}

// ---------------------------------------------------------------------------
extern "C" void kernel_launch(void* const* d_in, const int* in_sizes, int n_in,
                              void* d_out, int out_size, void* d_ws,
                              size_t ws_size, hipStream_t stream) {
  const float* x    = (const float*)d_in[0];
  const float* Wih0 = (const float*)d_in[1];
  const float* Whh0 = (const float*)d_in[2];
  const float* bih0 = (const float*)d_in[3];
  const float* bhh0 = (const float*)d_in[4];
  const float* Wih  = (const float*)d_in[5];
  const float* Whh  = (const float*)d_in[6];
  const float* bih  = (const float*)d_in[7];
  const float* bhh  = (const float*)d_in[8];
  const float* Wfc  = (const float*)d_in[9];
  const float* bfc  = (const float*)d_in[10];

  _Float16* wsh = (_Float16*)d_ws;
  float* wsb = (float*)((char*)d_ws + (size_t)NHALF * 2);

  int prep_n = NHALF + NBIAS;
  rnn_prep<<<(prep_n + 255) / 256, 256, 0, stream>>>(
      Wih0, Whh0, bih0, bhh0, Wih, Whh, bih, bhh, wsh, wsb);

  size_t shmem = (size_t)SMEM_HALVES * sizeof(_Float16);
  rnn_fused<<<RNN_B / BT, 256, shmem, stream>>>(x, wsh, wsb, Wfc, bfc,
                                                (float*)d_out);
}
